// ProgSGStyleModel_37434934952303
// MI455X (gfx1250) — matrix operand order repaired and checked by reading, not compiled, not run on hardware
//
#include <hip/hip_runtime.h>
#include <hip/hip_bf16.h>

// ---------------------------------------------------------------------------
// ProgSG-style graph transformer forward, CDNA5 (gfx1250, wave32) version.
// GEMMs run on v_wmma_f32_16x16x32_bf16 (bf16 in, f32 accumulate) with
// fragment-order pre-packed weights (B loads = global_load_b128).
// ---------------------------------------------------------------------------

#define HID 256
#define NHEAD 4
#define CHD 64

typedef __attribute__((ext_vector_type(16))) __bf16 v16bf;
typedef __attribute__((ext_vector_type(8)))  float  v8f;

union FragA { v16bf v; uint4 q[2]; };

static __device__ __forceinline__ unsigned enc_f32(float f) {
    unsigned u = __float_as_uint(f);
    return (u & 0x80000000u) ? ~u : (u | 0x80000000u);
}
static __device__ __forceinline__ float dec_f32(unsigned u) {
    u = (u & 0x80000000u) ? (u & 0x7fffffffu) : ~u;
    return __uint_as_float(u);
}

// ------------------------------- fills -------------------------------------
__global__ void fill_f32_kernel(float* p, float v, long long n) {
    long long i = blockIdx.x * (long long)blockDim.x + threadIdx.x;
    if (i < n) p[i] = v;
}
__global__ void fill_u32_kernel(unsigned* p, unsigned v, long long n) {
    long long i = blockIdx.x * (long long)blockDim.x + threadIdx.x;
    if (i < n) p[i] = v;
}

// -------------------- weight repack: f32 [K=256][N=256] -> fragment order --
// Packed layout: [mat][ntile(16)][kstep(8)][lane(32)][elem(16)] bf16.
// elem i of a lane = W[kstep*32 + (lane>=16?8:0) + (i&7) + (i>=8?16:0)][ntile*16 + (lane&15)]
// (CDNA5 ISA 7.12.2 B-matrix 32x16 bf16 wave32 layout, K pairs per VGPR.)
__global__ void pack_w_kernel(const float* __restrict__ W, __bf16* __restrict__ Wp,
                              long long total) {
    long long id = blockIdx.x * (long long)blockDim.x + threadIdx.x;
    if (id >= total) return;
    long long mat = id >> 16;
    int t     = (int)(id & 65535);
    int i     = t & 15;
    int lane  = (t >> 4) & 31;
    int ks    = (t >> 9) & 7;
    int ntile = (t >> 12) & 15;
    int nn = ntile * 16 + (lane & 15);
    int kk = ks * 32 + ((lane >> 4) * 8) + (i & 7) + ((i >> 3) * 16);
    Wp[id] = (__bf16)W[mat * (HID * HID) + (size_t)kk * HID + nn];
}

// --------------------------- embeddings ------------------------------------
__global__ void embed_nodes_kernel(const int* __restrict__ x,
                                   const float* __restrict__ e0,
                                   const float* __restrict__ e1,
                                   const float* __restrict__ e2,
                                   float* __restrict__ h,
                                   __bf16* __restrict__ hbf, int N) {
    long long id = blockIdx.x * (long long)blockDim.x + threadIdx.x;
    if (id >= (long long)N * HID) return;
    int c = (int)(id & (HID - 1));
    int n = (int)(id >> 8);
    float v = e0[x[n * 3 + 0] * HID + c] + e1[x[n * 3 + 1] * HID + c] + e2[x[n * 3 + 2] * HID + c];
    h[id]   = v;
    hbf[id] = (__bf16)v;
}

__global__ void embed_edges_kernel(const int* __restrict__ ea,
                                   const float* __restrict__ e0,
                                   const float* __restrict__ e1,
                                   __bf16* __restrict__ efbf, long long E) {
    long long id = blockIdx.x * (long long)blockDim.x + threadIdx.x;
    if (id >= E * HID) return;
    int c = (int)(id & (HID - 1));
    long long e = id >> 8;
    float v = e0[ea[e * 2 + 0] * HID + c] + e1[ea[e * 2 + 1] * HID + c];
    efbf[id] = (__bf16)v;
}

// ------------------------------ WMMA GEMM ----------------------------------
// Y[M,256] = X[M,256] @ W[256,256] + bias.
// Block = 512 threads = 16 waves. Block tile: 64 rows x 256 cols.
// Each wave owns one 16-wide N tile and 4 M tiles (4 accumulators); each
// B fragment (one 32B v16bf load from packed weights) feeds 4 WMMAs.
// All four A fragments are register-resident before the WMMA chain so the
// ds_load_b128s issue together and overlap the XDL pipeline.
template <bool BF16_OUT>
__global__ __launch_bounds__(512)
void wmma_gemm_bias_kernel(const __bf16* __restrict__ X,
                           const v16bf* __restrict__ Wp,   // packed fragments
                           const float* __restrict__ bias,
                           float* __restrict__ Yf,
                           __bf16* __restrict__ Ybf,
                           int M) {
    __shared__ __align__(16) __bf16 As[64 * HID];   // 32 KB

    const int row0 = blockIdx.x * 64;

    // cooperative A-tile load: 64 rows x 32 uint4 per row
    for (int i = threadIdx.x; i < 64 * 32; i += blockDim.x) {
        int r = i >> 5;
        uint4 val = make_uint4(0u, 0u, 0u, 0u);
        if (row0 + r < M)
            val = ((const uint4*)X)[(size_t)(row0 + r) * 16 + (i & 31)];
        ((uint4*)As)[i] = val;
    }
    __syncthreads();

    const int wave = threadIdx.x >> 5;      // N tile index 0..15
    const int lane = threadIdx.x & 31;
    const int col0 = wave * 16;

    const int mlane = lane & 15;            // A row within tile
    const int kgrp  = (lane >> 4) * 8;      // A K sub-offset (0 or 8)
    const int n     = lane & 15;            // D column
    const int mrow  = (lane >> 4) * 8;      // D row offset (0 or 8)

    v8f acc[4] = {{}, {}, {}, {}};

#pragma unroll
    for (int ks = 0; ks < 8; ++ks) {
        // one 32-byte fragment load per k-step, reused by 4 WMMAs
        v16bf b = Wp[((size_t)wave * 8 + ks) * 32 + lane];
        FragA a[4];
#pragma unroll
        for (int mt = 0; mt < 4; ++mt) {
            const __bf16* arow = As + (size_t)(mt * 16 + mlane) * HID + ks * 32 + kgrp;
            a[mt].q[0] = *(const uint4*)(arow);        // elems 0..7  : K = kgrp+0..7
            a[mt].q[1] = *(const uint4*)(arow + 16);   // elems 8..15 : K = kgrp+16..23
        }
#pragma unroll
        for (int mt = 0; mt < 4; ++mt) {
            acc[mt] = __builtin_amdgcn_wmma_f32_16x16x32_bf16(
                false, a[mt].v, false, b, (short)0, acc[mt], false, false);
        }
    }

    float bcol = bias[col0 + n];
#pragma unroll
    for (int mt = 0; mt < 4; ++mt) {
#pragma unroll
        for (int r = 0; r < 8; ++r) {
            int gm = row0 + mt * 16 + mrow + r;
            if (gm < M) {
                float val = acc[mt][r] + bcol;
                size_t idx = (size_t)gm * HID + col0 + n;
                if (BF16_OUT) Ybf[idx] = (__bf16)val;
                else          Yf[idx]  = val;
            }
        }
    }
}

// ------------------------- attention (edges) -------------------------------
__global__ void edge_alpha_kernel(const float* __restrict__ q,
                                  const float* __restrict__ k,
                                  const __bf16* __restrict__ e,
                                  const int* __restrict__ src,
                                  const int* __restrict__ dst,
                                  float* __restrict__ alpha,
                                  unsigned* __restrict__ amax_enc, long long E) {
    long long id = blockIdx.x * (long long)blockDim.x + threadIdx.x;
    if (id >= E * NHEAD) return;
    int hh = (int)(id & 3);
    long long edge = id >> 2;
    int sN = src[edge], dN = dst[edge];
    const float*  qp = q + (size_t)dN * HID + hh * CHD;
    const float*  kp = k + (size_t)sN * HID + hh * CHD;
    const __bf16* ep = e + (size_t)edge * HID + hh * CHD;
    float s = 0.f;
#pragma unroll 8
    for (int c = 0; c < CHD; ++c) s += qp[c] * (kp[c] + (float)ep[c]);
    s *= 0.125f;                                   // 1/sqrt(64)
    alpha[id] = s;
    atomicMax(&amax_enc[dN * NHEAD + hh], enc_f32(s));
}

__global__ void edge_exp_kernel(const float* __restrict__ alpha,
                                const unsigned* __restrict__ amax_enc,
                                const int* __restrict__ dst,
                                float* __restrict__ ea,
                                float* __restrict__ denom, long long E) {
    long long id = blockIdx.x * (long long)blockDim.x + threadIdx.x;
    if (id >= E * NHEAD) return;
    int hh = (int)(id & 3);
    long long edge = id >> 2;
    int dN = dst[edge];
    float m = dec_f32(amax_enc[dN * NHEAD + hh]);
    float v = __expf(alpha[id] - m);
    ea[id] = v;
    atomicAdd(&denom[dN * NHEAD + hh], v);
}

// 4 channels per thread: float4 value load, 8B bf16 edge load, 4 f32 atomics.
__global__ void edge_message_kernel(const float* __restrict__ v,
                                    const __bf16* __restrict__ e,
                                    const float* __restrict__ ea,
                                    const float* __restrict__ denom,
                                    const int* __restrict__ src,
                                    const int* __restrict__ dst,
                                    float* __restrict__ out_agg, long long E) {
    long long id = blockIdx.x * (long long)blockDim.x + threadIdx.x;
    if (id >= E * (HID / 4)) return;
    int c4 = (int)(id & (HID / 4 - 1)) * 4;       // channel base 0..252
    long long edge = id >> 6;
    int sN = src[edge], dN = dst[edge];
    int hh = c4 >> 6;
    float attn = ea[edge * NHEAD + hh] / fmaxf(denom[dN * NHEAD + hh], 1e-16f);
    const float*  vp = v + (size_t)sN * HID + c4;
    const __bf16* ep = e + (size_t)edge * HID + c4;
    float4 vv = *(const float4*)vp;
    float* op = out_agg + (size_t)dN * HID + c4;
    atomicAdd(op + 0, (vv.x + (float)ep[0]) * attn);
    atomicAdd(op + 1, (vv.y + (float)ep[1]) * attn);
    atomicAdd(op + 2, (vv.z + (float)ep[2]) * attn);
    atomicAdd(op + 3, (vv.w + (float)ep[3]) * attn);
}

// -------------------- gate + ReLU + LayerNorm + residual -------------------
// One wave (32 lanes) per node; each lane owns 8 channels.
__global__ __launch_bounds__(256)
void node_update_kernel(const float* __restrict__ out_agg,
                        const float* __restrict__ r,
                        const float* __restrict__ wbeta,   // [768]
                        const float* __restrict__ ln_g,
                        const float* __restrict__ ln_b,
                        float* __restrict__ h,
                        __bf16* __restrict__ hbf, int N) {
    int node = blockIdx.x * 8 + (threadIdx.x >> 5);
    int lane = threadIdx.x & 31;
    if (node >= N) return;
    const float* o  = out_agg + (size_t)node * HID;
    const float* rr = r       + (size_t)node * HID;

    float lo[8], lr[8];
    float s = 0.f;
#pragma unroll
    for (int j = 0; j < 8; ++j) {
        int c = lane * 8 + j;
        float ov = o[c], rv = rr[c];
        lo[j] = ov; lr[j] = rv;
        s += ov * wbeta[c] + rv * wbeta[HID + c] + (ov - rv) * wbeta[2 * HID + c];
    }
#pragma unroll
    for (int off = 16; off; off >>= 1) s += __shfl_xor(s, off, 32);
    float beta = 1.f / (1.f + __expf(-s));

    float g[8], sum = 0.f, sumsq = 0.f;
#pragma unroll
    for (int j = 0; j < 8; ++j) {
        float gv = fmaxf(beta * lr[j] + (1.f - beta) * lo[j], 0.f);
        g[j] = gv; sum += gv; sumsq += gv * gv;
    }
#pragma unroll
    for (int off = 16; off; off >>= 1) {
        sum   += __shfl_xor(sum,   off, 32);
        sumsq += __shfl_xor(sumsq, off, 32);
    }
    float mu  = sum * (1.f / HID);
    float var = sumsq * (1.f / HID) - mu * mu;
    float inv = rsqrtf(var + 1e-5f);
#pragma unroll
    for (int j = 0; j < 8; ++j) {
        int c = lane * 8 + j;
        float ln = (g[j] - mu) * inv * ln_g[c] + ln_b[c];
        size_t idx = (size_t)node * HID + c;
        float hv = h[idx] + ln;
        h[idx]   = hv;
        hbf[idx] = (__bf16)hv;
    }
}

// ------------------------------- pooling -----------------------------------
// 4 channels per thread.
__global__ void pool_kernel(const float* __restrict__ h,
                            const int* __restrict__ batch,
                            float* __restrict__ pooled, int N) {
    long long id = blockIdx.x * (long long)blockDim.x + threadIdx.x;
    if (id >= (long long)N * (HID / 4)) return;
    int c4 = (int)(id & (HID / 4 - 1)) * 4;
    int n  = (int)(id >> 6);
    float4 hv = *(const float4*)(h + (size_t)n * HID + c4);
    float* pp = pooled + (size_t)batch[n] * HID + c4;
    atomicAdd(pp + 0, hv.x);
    atomicAdd(pp + 1, hv.y);
    atomicAdd(pp + 2, hv.z);
    atomicAdd(pp + 3, hv.w);
}

// ------------------------------- readout -----------------------------------
__global__ __launch_bounds__(256)
void readout_kernel(const float* __restrict__ pooled,
                    const float* __restrict__ scalars,
                    const float* __restrict__ pw1, const float* __restrict__ pb1,
                    const float* __restrict__ pw2, const float* __restrict__ pb2,
                    const float* __restrict__ rw1, const float* __restrict__ rb1,
                    const float* __restrict__ rw2, const float* __restrict__ rb2,
                    const float* __restrict__ rw3, const float* __restrict__ rb3,
                    float* __restrict__ outp) {
    __shared__ float zA[HID], zB[HID], z2[HID / 2], sc[5];
    int g = blockIdx.x, t = threadIdx.x;
    if (t < 5) sc[t] = scalars[g * 5 + t];
    __syncthreads();

    float p1 = pb1[t];
    for (int i = 0; i < 5; ++i) p1 += sc[i] * pw1[i * HID + t];
    zA[t] = fmaxf(p1, 0.f);
    __syncthreads();

    float p2 = pb2[t];
    for (int i = 0; i < HID; ++i) p2 += zA[i] * pw2[i * HID + t];
    float pg = pooled[(size_t)g * HID + t] + p2;
    __syncthreads();
    zB[t] = pg;
    __syncthreads();

    float a1 = rb1[t];
    for (int i = 0; i < HID; ++i) a1 += zB[i] * rw1[i * HID + t];
    zA[t] = fmaxf(a1, 0.f);
    __syncthreads();

    if (t < HID / 2) {
        float a2 = rb2[t];
        for (int i = 0; i < HID; ++i) a2 += zA[i] * rw2[i * (HID / 2) + t];
        z2[t] = fmaxf(a2, 0.f);
    }
    __syncthreads();

    if (t < 4) {
        float a3 = rb3[t];
        for (int i = 0; i < HID / 2; ++i) a3 += z2[i] * rw3[i * 4 + t];
        outp[g * 4 + t] = a3;
    }
}

// ---------------------------------------------------------------------------
static inline long long cdiv(long long a, long long b) { return (a + b - 1) / b; }

extern "C" void kernel_launch(void* const* d_in, const int* in_sizes, int n_in,
                              void* d_out, int out_size, void* d_ws, size_t ws_size,
                              hipStream_t stream) {
    const int*   x         = (const int*)d_in[0];
    const int*   edge_attr = (const int*)d_in[1];
    const int*   edge_idx  = (const int*)d_in[2];
    const int*   batch     = (const int*)d_in[3];
    const float* scalars   = (const float*)d_in[4];
    const float* node_emb0 = (const float*)d_in[6];
    const float* node_emb1 = (const float*)d_in[7];
    const float* node_emb2 = (const float*)d_in[8];
    const float* edge_emb0 = (const float*)d_in[9];
    const float* edge_emb1 = (const float*)d_in[10];
    const float* Wq = (const float*)d_in[11]; const float* bq = (const float*)d_in[12];
    const float* Wk = (const float*)d_in[13]; const float* bk = (const float*)d_in[14];
    const float* Wv = (const float*)d_in[15]; const float* bv = (const float*)d_in[16];
    const float* We = (const float*)d_in[17]; const float* be = (const float*)d_in[18];
    const float* Ws = (const float*)d_in[19]; const float* bs = (const float*)d_in[20];
    const float* Wbeta = (const float*)d_in[21];
    const float* ln_g  = (const float*)d_in[22];
    const float* ln_b  = (const float*)d_in[23];
    const float* pw1 = (const float*)d_in[24]; const float* pb1 = (const float*)d_in[25];
    const float* pw2 = (const float*)d_in[26]; const float* pb2 = (const float*)d_in[27];
    const float* rw1 = (const float*)d_in[28]; const float* rb1 = (const float*)d_in[29];
    const float* rw2 = (const float*)d_in[30]; const float* rb2 = (const float*)d_in[31];
    const float* rw3 = (const float*)d_in[32]; const float* rb3 = (const float*)d_in[33];
    float* outp = (float*)d_out;

    const int       N = in_sizes[0] / 3;
    const long long E = in_sizes[1] / 2;
    const int       G = in_sizes[4] / 5;
    const int       L = in_sizes[11] / (HID * HID);
    const int* src = edge_idx;
    const int* dst = edge_idx + E;

    // ---- workspace carve ----
    char* wpc = (char*)d_ws;
    auto carve = [&](size_t bytes) -> void* {
        void* r = (void*)wpc;
        wpc += (bytes + 255) & ~(size_t)255;
        return r;
    };
    float*   h       = (float*)  carve((size_t)N * HID * 4);
    __bf16*  hbf     = (__bf16*) carve((size_t)N * HID * 2);
    __bf16*  efbf    = (__bf16*) carve((size_t)E * HID * 2);
    float*   qb      = (float*)  carve((size_t)N * HID * 4);
    float*   kb_     = (float*)  carve((size_t)N * HID * 4);
    float*   vb      = (float*)  carve((size_t)N * HID * 4);
    float*   rb_     = (float*)  carve((size_t)N * HID * 4);
    __bf16*  ebf     = (__bf16*) carve((size_t)E * HID * 2);
    float*   alpha   = (float*)  carve((size_t)E * NHEAD * 4);
    float*   eav     = (float*)  carve((size_t)E * NHEAD * 4);
    unsigned* amaxE  = (unsigned*)carve((size_t)N * NHEAD * 4);
    float*   denom   = (float*)  carve((size_t)N * NHEAD * 4);
    float*   out_agg = (float*)  carve((size_t)N * HID * 4);
    float*   pooled  = (float*)  carve((size_t)G * HID * 4);
    __bf16*  wq_bf   = (__bf16*) carve((size_t)L * HID * HID * 2);
    __bf16*  wk_bf   = (__bf16*) carve((size_t)L * HID * HID * 2);
    __bf16*  wv_bf   = (__bf16*) carve((size_t)L * HID * HID * 2);
    __bf16*  we_bf   = (__bf16*) carve((size_t)L * HID * HID * 2);
    __bf16*  ws_bf   = (__bf16*) carve((size_t)L * HID * HID * 2);

    const int TB = 256;
    const long long wTot = (long long)L * HID * HID;

    // weights -> fragment-order bf16 (once per call)
    pack_w_kernel<<<cdiv(wTot, TB), TB, 0, stream>>>(Wq, wq_bf, wTot);
    pack_w_kernel<<<cdiv(wTot, TB), TB, 0, stream>>>(Wk, wk_bf, wTot);
    pack_w_kernel<<<cdiv(wTot, TB), TB, 0, stream>>>(Wv, wv_bf, wTot);
    pack_w_kernel<<<cdiv(wTot, TB), TB, 0, stream>>>(We, we_bf, wTot);
    pack_w_kernel<<<cdiv(wTot, TB), TB, 0, stream>>>(Ws, ws_bf, wTot);

    // embeddings
    embed_nodes_kernel<<<cdiv((long long)N * HID, TB), TB, 0, stream>>>(
        x, node_emb0, node_emb1, node_emb2, h, hbf, N);
    embed_edges_kernel<<<cdiv(E * HID, TB), TB, 0, stream>>>(
        edge_attr, edge_emb0, edge_emb1, efbf, E);

    const int nTilesN = (int)cdiv(N, 64);
    const int nTilesE = (int)cdiv(E, 64);

    for (int l = 0; l < L; ++l) {
        const size_t wo = (size_t)l * HID * HID;
        const size_t bo = (size_t)l * HID;

        wmma_gemm_bias_kernel<false><<<nTilesN, 512, 0, stream>>>(
            hbf, (const v16bf*)(wq_bf + wo), bq + bo, qb, nullptr, N);
        wmma_gemm_bias_kernel<false><<<nTilesN, 512, 0, stream>>>(
            hbf, (const v16bf*)(wk_bf + wo), bk + bo, kb_, nullptr, N);
        wmma_gemm_bias_kernel<false><<<nTilesN, 512, 0, stream>>>(
            hbf, (const v16bf*)(wv_bf + wo), bv + bo, vb, nullptr, N);
        wmma_gemm_bias_kernel<false><<<nTilesN, 512, 0, stream>>>(
            hbf, (const v16bf*)(ws_bf + wo), bs + bo, rb_, nullptr, N);
        wmma_gemm_bias_kernel<true><<<nTilesE, 512, 0, stream>>>(
            efbf, (const v16bf*)(we_bf + wo), be + bo, nullptr, ebf, (int)E);

        fill_u32_kernel<<<cdiv((long long)N * NHEAD, TB), TB, 0, stream>>>(amaxE, 0u, (long long)N * NHEAD);
        fill_f32_kernel<<<cdiv((long long)N * NHEAD, TB), TB, 0, stream>>>(denom, 0.f, (long long)N * NHEAD);
        fill_f32_kernel<<<cdiv((long long)N * HID, TB), TB, 0, stream>>>(out_agg, 0.f, (long long)N * HID);

        edge_alpha_kernel<<<cdiv(E * NHEAD, TB), TB, 0, stream>>>(
            qb, kb_, ebf, src, dst, alpha, amaxE, E);
        edge_exp_kernel<<<cdiv(E * NHEAD, TB), TB, 0, stream>>>(
            alpha, amaxE, dst, eav, denom, E);
        edge_message_kernel<<<cdiv(E * (HID / 4), TB), TB, 0, stream>>>(
            vb, ebf, eav, denom, src, dst, out_agg, E);

        node_update_kernel<<<cdiv(N, 8), 256, 0, stream>>>(
            out_agg, rb_, Wbeta + (size_t)l * 3 * HID, ln_g + bo, ln_b + bo, h, hbf, N);
    }

    fill_f32_kernel<<<cdiv((long long)G * HID, TB), TB, 0, stream>>>(pooled, 0.f, (long long)G * HID);
    pool_kernel<<<cdiv((long long)N * (HID / 4), TB), TB, 0, stream>>>(h, batch, pooled, N);

    readout_kernel<<<G, 256, 0, stream>>>(pooled, scalars,
                                          pw1, pb1, pw2, pb2,
                                          rw1, rb1, rw2, rb2, rw3, rb3, outp);
}
